// DataEmbedding_inverted_LSTM_60627758350808
// MI455X (gfx1250) — matrix operand order repaired and checked by reading, not compiled, not run on hardware
//
#include <hip/hip_runtime.h>
#include <hip/hip_bf16.h>

typedef __attribute__((ext_vector_type(16))) _Float16 v16h;
typedef __attribute__((ext_vector_type(8)))  float    v8f;

#define BATCH  1024
#define SEQ    512
#define CIN    46
#define HID    16
#define DMODEL 512

__device__ __forceinline__ float fast_sigmoid(float x) {
    return __builtin_amdgcn_rcpf(1.0f + __expf(-x));
}
__device__ __forceinline__ float fast_tanh(float x) {
    // tanh(x) = 2*sigmoid(2x) - 1
    return 2.0f * fast_sigmoid(2.0f * x) - 1.0f;
}

// ---------------------------------------------------------------------------
// Prep 1: shuffle ve_W [D=512][T=512] (f32) into WMMA B-fragment order (f16):
//   vefrag[((kt*32 + gnt)*32 + lane)*16 + u]  with  K = kt*32 + (u%8)+16*(u/8)+8*(lane/16),
//   N = gnt*16 + lane%16,  element = ve_W[N][K]   (B[t][d] = ve_W[d][t])
// ---------------------------------------------------------------------------
__global__ void prep_vefrag_kernel(const float* __restrict__ veW,
                                   _Float16* __restrict__ vefrag) {
    int idx = blockIdx.x * blockDim.x + threadIdx.x;   // 0 .. 262143
    int u    = idx & 15;
    int lane = (idx >> 4) & 31;
    int gnt  = (idx >> 9) & 31;
    int kt   = idx >> 14;
    int K = kt * 32 + ((lane >> 4) << 3) + (u & 7) + ((u >> 3) << 4);
    int N = gnt * 16 + (lane & 15);
    vefrag[idx] = (_Float16)veW[(size_t)N * SEQ + K];
}

// Prep 2: S[d] = sum_t ve_W[d][t]  (carries fc bias through the big GEMM)
__global__ void prep_ssum_kernel(const float* __restrict__ veW,
                                 float* __restrict__ S) {
    int d = blockIdx.x * blockDim.x + threadIdx.x;
    if (d < DMODEL) {
        float s = 0.0f;
        for (int t = 0; t < SEQ; ++t) s += veW[(size_t)d * SEQ + t];
        S[d] = s;
    }
}

// ---------------------------------------------------------------------------
// Fused input-projection + LSTM scan, software-pipelined 2 steps deep.
// One wave per 16 batch rows.  Per step:
//   gates[16x64] = [x(46)|h(16)|0 0] (A, via double-buffered LDS)
//                  @ [W_ih^T ; W_hh^T] (B, resident in VGPRs)
// as 8x v_wmma_f32_16x16x32_f16.  Critical path per step is only
// h-store -> s_wait_dscnt -> A-load -> WMMA -> exp/rcp; global x loads for
// t+2 are issued here and land 1.5 iterations later.
// ---------------------------------------------------------------------------
__device__ __forceinline__ void lstm_step(
    int t, const float* __restrict__ x,
    _Float16* __restrict__ cur, _Float16* __restrict__ nxt,
    const int* goff, const int* loff,
    float* xr_st,     // holds x(t+1), staged into `nxt` this step
    float* xr_ld,     // receives x(t+2) global loads this step
    const v16h* bfr, const float* bias,
    v8f& cst, _Float16* __restrict__ afrag_b0,
    int lhi, int j)
{
    // previous step's h stores + previous step's x staging are now required
    asm volatile("s_wait_dscnt 0" ::: "memory");

    // A fragments for step t (16-bit A layout: M=L%16, K=(u%8)+16*(u/8)+8*(L/16))
    v16h a0, a1;
    #pragma unroll
    for (int u = 0; u < 16; ++u) {
        int Kb = (u & 7) + ((u >> 3) << 4) + (lhi << 3);
        a0[u] = cur[j * 64 + Kb];
        a1[u] = cur[j * 64 + 32 + Kb];
    }

    // issue global loads of x(t+2) (off the critical path; used next step)
    if (t + 2 < SEQ) {
        #pragma unroll
        for (int k = 0; k < 23; ++k) xr_ld[k] = x[goff[k] + (t + 2) * CIN];
    }

    v8f g[4];
    #pragma unroll
    for (int nt = 0; nt < 4; ++nt) {
        v8f c0;
        #pragma unroll
        for (int r = 0; r < 8; ++r) c0[r] = bias[nt];
        c0 = __builtin_amdgcn_wmma_f32_16x16x32_f16(
                 false, a0, false, bfr[nt],     (short)0, c0, false, false);
        c0 = __builtin_amdgcn_wmma_f32_16x16x32_f16(
                 false, a1, false, bfr[4 + nt], (short)0, c0, false, false);
        g[nt] = c0;
    }

    // stage x(t+1) into the other buffer (loaded during previous step)
    if (t + 1 < SEQ) {
        #pragma unroll
        for (int k = 0; k < 23; ++k) nxt[loff[k]] = (_Float16)xr_st[k];
    }

    // lane-local LSTM cell: gate n%16 == hidden unit j for all 4 gate groups
    const int kt_t = t >> 5, kk = t & 31;
    const int gsel = (kk >> 3) & 1;
    const int u_t  = (kk & 7) + ((kk >> 4) << 3);
    #pragma unroll
    for (int r = 0; r < 8; ++r) {
        float iv = fast_sigmoid(g[0][r]);
        float fv = fast_sigmoid(g[1][r]);
        float gv = fast_tanh(g[2][r]);
        float ov = fast_sigmoid(g[3][r]);
        float cv = fv * cst[r] + iv * gv;
        cst[r] = cv;
        float hv = ov * fast_tanh(cv);
        int M = r + 8 * lhi;                      // batch row in tile
        nxt[M * 64 + CIN + j] = (_Float16)hv;     // feed step t+1
        // phase-3 A-fragment order: lane' = gsel*16 + j, half = u_t  => K == t
        afrag_b0[((M * 16 + kt_t) * 32 + (gsel * 16 + j)) * 16 + u_t] = (_Float16)hv;
    }
}

__global__ void __launch_bounds__(32) lstm_kernel(
    const float* __restrict__ x,   const float* __restrict__ Wih,
    const float* __restrict__ Whh, const float* __restrict__ bih,
    const float* __restrict__ bhh, _Float16* __restrict__ afrag) {
    __shared__ _Float16 xh[2][16][64];       // double-buffered [row][K column]
    const int lane = threadIdx.x & 31;
    const int b0   = blockIdx.x * 16;
    const int lhi  = lane >> 4;              // lane group 0/1
    const int j    = lane & 15;

    // --- constant B fragments: Wcat[K][N], K:0..45=W_ih^T, 46..61=W_hh^T ---
    v16h bfr[8];                             // [kt*4 + nt]
    #pragma unroll
    for (int kt = 0; kt < 2; ++kt) {
        #pragma unroll
        for (int nt = 0; nt < 4; ++nt) {
            v16h v;
            #pragma unroll
            for (int u = 0; u < 16; ++u) {
                int K = kt * 32 + lhi * 8 + (u & 7) + ((u >> 3) << 4);      // B-frag: K = (u%8)+16*(u/8)+8*(L/16)
                int N = nt * 16 + j;                 //         N = L%16
                float w = 0.0f;
                if (K < CIN)            w = Wih[N * CIN + K];
                else if (K < CIN + HID) w = Whh[N * HID + (K - CIN)];
                v[u] = (_Float16)w;
            }
            bfr[kt * 4 + nt] = v;
        }
    }
    float bias[4];
    #pragma unroll
    for (int nt = 0; nt < 4; ++nt) {
        int n = nt * 16 + j;
        bias[nt] = bih[n] + bhh[n];
    }

    // per-lane x-tile gather offsets: 16*46 = 32*23 elements, exact split
    int goff[23], loff[23];
    #pragma unroll
    for (int k = 0; k < 23; ++k) {
        int i = lane + 32 * k;
        int row = i / CIN, col = i - row * CIN;
        goff[k] = (b0 + row) * (SEQ * CIN) + col;
        loff[k] = row * 64 + col;
    }

    // zero both buffers (h cols 46..61 and pad 62..63 start at 0)
    for (int i = lane; i < 2 * 16 * 64; i += 32)
        ((_Float16*)&xh[0][0][0])[i] = (_Float16)0.0f;

    _Float16* buf0 = &xh[0][0][0];
    _Float16* buf1 = &xh[1][0][0];

    // prologue: x(0) -> buf0, x(1) -> xrA
    float xrA[23], xrB[23];
    #pragma unroll
    for (int k = 0; k < 23; ++k) xrA[k] = x[goff[k]];
    #pragma unroll
    for (int k = 0; k < 23; ++k) buf0[loff[k]] = (_Float16)xrA[k];
    #pragma unroll
    for (int k = 0; k < 23; ++k) xrA[k] = x[goff[k] + CIN];

    v8f cst = {};                            // c state: rows r+8*lhi, hidden j
    _Float16* afrag_b0 = afrag + (size_t)b0 * (16 * 32 * 16);

    for (int t = 0; t < SEQ; t += 2) {
        lstm_step(t,     x, buf0, buf1, goff, loff, xrA, xrB, bfr, bias, cst,
                  afrag_b0, lhi, j);
        lstm_step(t + 1, x, buf1, buf0, goff, loff, xrB, xrA, bfr, bias, cst,
                  afrag_b0, lhi, j);
    }
}

// ---------------------------------------------------------------------------
// out[b] = fc_W @ (h[b]^T @ ve_W^T) + fc_b (x) S + ve_b
// Z GEMM: M=16(hidden), N=512(d_model), K=512(t) per batch, f16 WMMA.
// 1024 blocks (one batch each) x 8 waves (64 d-columns each).
// ---------------------------------------------------------------------------
__global__ void __launch_bounds__(256) embed_kernel(
    const _Float16* __restrict__ afrag, const _Float16* __restrict__ vefrag,
    const float* __restrict__ Ssum,     const float* __restrict__ fcW,
    const float* __restrict__ fcb,      const float* __restrict__ veb,
    float* __restrict__ out) {
    __shared__ float Zs[8][16][65];               // pad 65 -> conflict-free cols
    const int lane = threadIdx.x & 31;
    const int wave = threadIdx.x >> 5;
    const int b    = blockIdx.x;
    const int lhi  = lane >> 4, j = lane & 15;

    v8f acc[4] = {{}, {}, {}, {}};
    for (int kt = 0; kt < 16; ++kt) {
        v16h a = *(const v16h*)(afrag + (((size_t)b * 16 + kt) * 32 + lane) * 16);
        #pragma unroll
        for (int nt = 0; nt < 4; ++nt) {
            int gnt = wave * 4 + nt;              // global 16-wide d-tile
            v16h bv = *(const v16h*)(vefrag + (((size_t)kt * 32 + gnt) * 32 + lane) * 16);
            acc[nt] = __builtin_amdgcn_wmma_f32_16x16x32_f16(
                          false, a, false, bv, (short)0, acc[nt], false, false);
        }
    }

    // dump Z (D layout: M = r + 8*(lane/16) = hidden k, N = lane%16) to LDS
    #pragma unroll
    for (int nt = 0; nt < 4; ++nt)
        #pragma unroll
        for (int r = 0; r < 8; ++r)
            Zs[wave][r + 8 * lhi][nt * 16 + j] = acc[nt][r];
    asm volatile("s_wait_dscnt 0" ::: "memory");

    // fc epilogue: each lane owns d-columns (lane, lane+32) of this wave's 64
    const int n0 = wave * 64;
    float z0[16], z1[16];
    #pragma unroll
    for (int k = 0; k < 16; ++k) {
        z0[k] = Zs[wave][k][lane];
        z1[k] = Zs[wave][k][lane + 32];
    }
    const int d0 = n0 + lane, d1 = n0 + lane + 32;
    const float s0 = Ssum[d0], s1 = Ssum[d1];
    const float vb0 = veb[d0], vb1 = veb[d1];
    float* outb = out + (size_t)b * CIN * DMODEL;
    for (int c = 0; c < CIN; ++c) {
        float fb = fcb[c];
        float a0 = fb * s0 + vb0, a1 = fb * s1 + vb1;
        #pragma unroll
        for (int k = 0; k < 16; ++k) {
            float w = fcW[c * HID + k];           // uniform -> scalar loads
            a0 += w * z0[k];
            a1 += w * z1[k];
        }
        outb[(size_t)c * DMODEL + d0] = a0;
        outb[(size_t)c * DMODEL + d1] = a1;
    }
}

extern "C" void kernel_launch(void* const* d_in, const int* in_sizes, int n_in,
                              void* d_out, int out_size, void* d_ws, size_t ws_size,
                              hipStream_t stream) {
    const float* x    = (const float*)d_in[0];
    // d_in[1] = x_mark (unused by reference forward)
    const float* Wih  = (const float*)d_in[2];
    const float* Whh  = (const float*)d_in[3];
    const float* bih  = (const float*)d_in[4];
    const float* bhh  = (const float*)d_in[5];
    const float* fcW  = (const float*)d_in[6];
    const float* fcb  = (const float*)d_in[7];
    const float* veW  = (const float*)d_in[8];
    const float* veb  = (const float*)d_in[9];
    float* out = (float*)d_out;

    char* ws = (char*)d_ws;
    _Float16* afrag  = (_Float16*)ws;                                   // 16 MB
    _Float16* vefrag = (_Float16*)(ws + (size_t)16 * 1024 * 1024);      // 512 KB
    float*    Ssum   = (float*)(ws + (size_t)16 * 1024 * 1024 + 512 * 1024); // 2 KB

    prep_vefrag_kernel<<<1024, 256, 0, stream>>>(veW, vefrag);
    prep_ssum_kernel  <<<2,    256, 0, stream>>>(veW, Ssum);
    lstm_kernel       <<<64,    32, 0, stream>>>(x, Wih, Whh, bih, bhh, afrag);
    embed_kernel      <<<1024, 256, 0, stream>>>(afrag, vefrag, Ssum, fcW, fcb, veb, out);
}
